// WindowAttentionNone_61675730371235
// MI455X (gfx1250) — matrix operand order repaired and checked by reading, not compiled, not run on hardware
//
#include <hip/hip_runtime.h>
#include <hip/hip_bf16.h>

typedef __attribute__((ext_vector_type(16))) _Float16 v16h;
typedef __attribute__((ext_vector_type(8)))  _Float16 v8h;
typedef __attribute__((ext_vector_type(8)))  float    v8f;
typedef __attribute__((ext_vector_type(4)))  unsigned v4u;
typedef __attribute__((ext_vector_type(8)))  int      v8i;
typedef __attribute__((ext_vector_type(4)))  int      v4i;

#define NTOK   64
#define CDIM   256
#define GAIN   0.0625f                 // 1/sqrt(256)
#define SCALEF 0.17677669529663689f    // 32^-0.5

// LDS layout (bytes)
#define OFF_XH   0                     // x as f16        [64][256]  32KB
#define OFF_NXH  32768                 // norm_x as f16   [64][256]  32KB
#define OFF_Q    65536                 // transient x f32 (64KB over Q+K); then Q [8][64][32]; then O [64][256]
#define OFF_K    98304                 // K [8][64][32]   32KB
#define OFF_VT   131072                // V^T [8][32][64] 32KB
#define OFF_WS   163840                // per-wave staging: 8 x 16KB (weights / attn probs)
#define SMEM_BYTES (163840 + 8 * 16384)   // 288 KB

#define WMAT_HALFS 65536               // 256*256 per matrix
#define WS_NEEDED  (4ull * WMAT_HALFS * sizeof(_Float16))

#if __has_builtin(__builtin_amdgcn_tensor_load_to_lds) && __has_builtin(__builtin_amdgcn_s_wait_tensorcnt)
#define HAVE_TDM 1
#else
#define HAVE_TDM 0
#endif

static __device__ __forceinline__ v16h ld_pair(const _Float16* p0, const _Float16* p1) {
  union { v16h v; v8h h[2]; } u;
  u.h[0] = *(const v8h*)p0;
  u.h[1] = *(const v8h*)p1;
  return u.v;
}
// A-matrix 16x32 f16 fragment: lane half g holds K in {g*8..g*8+7} U {16+g*8..23+g*8}
static __device__ __forceinline__ v16h ldA(const _Float16* row, int kb, int g) {
  return ld_pair(row + kb + g * 8, row + kb + 16 + g * 8);
}
// B-matrix 32x16 f16 fragment: lane half g holds K in {g*8..g*8+7} U {16+g*8..23+g*8}
static __device__ __forceinline__ v16h ldB(const _Float16* row, int kb, int g) {
  return ld_pair(row + kb + g * 8, row + kb + 16 + g * 8);
}

// Stage a 32x256 weight slice into per-wave LDS (f16, gain folded)
template<bool PREW>
static __device__ __forceinline__ void stage_weights(
    const float* __restrict__ Wg, const _Float16* __restrict__ Wh,
    _Float16* __restrict__ wsl, int w, int lane)
{
  if (PREW) {
    const _Float16* wsrc = Wh + (size_t)(w * 32) * 256;
    for (int t = lane; t < 1024; t += 32)
      *(v8h*)(wsl + t * 8) = *(const v8h*)(wsrc + t * 8);
  } else {
    for (int t = lane; t < 2048; t += 32) {
      const int row = t >> 6;
      const int col = (t & 63) << 2;
      const float4 wv = *(const float4*)(Wg + (size_t)(w * 32 + row) * 256 + col);
      wsl[row * 256 + col + 0] = (_Float16)(wv.x * GAIN);
      wsl[row * 256 + col + 1] = (_Float16)(wv.y * GAIN);
      wsl[row * 256 + col + 2] = (_Float16)(wv.z * GAIN);
      wsl[row * 256 + col + 3] = (_Float16)(wv.w * GAIN);
    }
  }
}

// dst(64 x 32) = src(64 x 256) @ (W[w*32..w*32+32)][:] * gain)^T + bias
template<bool PREW>
static __device__ __forceinline__ void project_slice(
    const _Float16* __restrict__ src, const float* __restrict__ Wg,
    const _Float16* __restrict__ Wh, const float* __restrict__ bias,
    _Float16* __restrict__ wsl, _Float16* __restrict__ dst,
    int w, int lane, bool transposed)
{
  const int lr = lane & 15, g = lane >> 4;
  stage_weights<PREW>(Wg, Wh, wsl, w, lane);
  // same-wave DS ops are in-order: no barrier needed for private slice
  for (int mt = 0; mt < 4; ++mt) {
    v16h A[8];
    const _Float16* arow = src + (mt * 16 + lr) * 256;
#pragma unroll
    for (int kk = 0; kk < 8; ++kk) A[kk] = ldA(arow, kk * 32, g);
#pragma unroll
    for (int nt = 0; nt < 2; ++nt) {
      v8f acc = {};
      const _Float16* brow = wsl + (nt * 16 + lr) * 256;
#pragma unroll
      for (int kk = 0; kk < 8; ++kk)
        acc = __builtin_amdgcn_wmma_f32_16x16x32_f16(false, A[kk], false,
                  ldB(brow, kk * 32, g), (short)0, acc, false, false);
      const float bv = bias[w * 32 + nt * 16 + lr];
#pragma unroll
      for (int i = 0; i < 8; ++i) {
        const int row = mt * 16 + i + g * 8;
        const int c   = nt * 16 + lr;
        const _Float16 val = (_Float16)(acc[i] + bv);
        if (transposed) dst[c * 64 + row] = val;   // V^T: [32][64]
        else            dst[row * 32 + c] = val;   // [64][32]
      }
    }
  }
}

// One-time f32 -> f16 (gain folded) conversion of the 4 weight matrices into d_ws
__global__ void __launch_bounds__(256)
prep_weights_kernel(const float* __restrict__ qw, const float* __restrict__ kw,
                    const float* __restrict__ vw, const float* __restrict__ pw,
                    _Float16* __restrict__ dst)
{
  const int idx = (blockIdx.x * 256 + threadIdx.x) * 4;   // 16384 threads * 4 = 65536
  const float* srcs[4] = { qw, kw, vw, pw };
#pragma unroll
  for (int m = 0; m < 4; ++m) {
    const float4 v = *(const float4*)(srcs[m] + idx);
    _Float16* o = dst + m * WMAT_HALFS + idx;
    o[0] = (_Float16)(v.x * GAIN);
    o[1] = (_Float16)(v.y * GAIN);
    o[2] = (_Float16)(v.z * GAIN);
    o[3] = (_Float16)(v.w * GAIN);
  }
}

template<bool PREW>
__global__ void __launch_bounds__(256, 1)
win_attn_kernel(const float* __restrict__ x,  const float* __restrict__ mask,
                const float* __restrict__ qw, const float* __restrict__ qb,
                const float* __restrict__ kw, const float* __restrict__ kbv,
                const float* __restrict__ vw, const float* __restrict__ vb,
                const float* __restrict__ pw, const float* __restrict__ pb,
                const _Float16* __restrict__ wf,
                float* __restrict__ out)
{
  extern __shared__ __align__(16) char smem[];
  _Float16* xh  = (_Float16*)(smem + OFF_XH);
  _Float16* nxh = (_Float16*)(smem + OFF_NXH);
  _Float16* qh  = (_Float16*)(smem + OFF_Q);
  _Float16* kh  = (_Float16*)(smem + OFF_K);
  _Float16* vt  = (_Float16*)(smem + OFF_VT);

  const int tid  = threadIdx.x;
  const int b    = blockIdx.x;
  const int w    = tid >> 5;     // wave id == head id
  const int lane = tid & 31;
  const int lr   = lane & 15, g = lane >> 4;
  _Float16* wsl  = (_Float16*)(smem + OFF_WS) + w * 8192;   // 16KB/wave

  const float* xb = x    + (size_t)b * NTOK * CDIM;
  const float* mk = mask + (size_t)b * NTOK * NTOK;

  // Warm L2 with the last-used weights (global_prefetch_b8)
  if (PREW) __builtin_prefetch(wf + 3 * WMAT_HALFS + (size_t)tid * 256, 0, 3);
  else      __builtin_prefetch(pw + (size_t)tid * 256, 0, 3);

  // ---- Phase 0: pull x tile (64x256 f32, 64KB) into LDS via the TDM ----
  const float* xf;
#if HAVE_TDM
  if (w == 0) {
    const unsigned long long ga = (unsigned long long)(uintptr_t)xb;
    // D# group 0: count=1 | lds_addr | global_addr | type=2
    v4u g0 = { 1u,
               (unsigned)OFF_Q,
               (unsigned)(ga & 0xffffffffu),
               (unsigned)((ga >> 32) & 0x01ffffffu) | (2u << 30) };
    // D# group 1: data_size=4B, tensor 256x64, tile 256x64, dim0 stride 256
    v8i g1 = { (int)0x20000,           // data_size=2 (4 bytes)
               (int)(256u << 16),      // tensor_dim0 = 256
               (int)(64u  << 16),      // tensor_dim1 = 64
               (int)(256u << 16),      // tile_dim0   = 256
               (int)64,                // tile_dim1   = 64
               (int)256,               // tensor_dim0_stride = 256
               (int)(16384u << 16),    // tensor_dim1_stride = 16384 (lo bits)
               0 };
    v4i z4 = { 0, 0, 0, 0 };
#if __clang_major__ >= 23
    v8i z8 = { 0, 0, 0, 0, 0, 0, 0, 0 };
    __builtin_amdgcn_tensor_load_to_lds(g0, g1, z4, z4, z8, 0);
#else
    __builtin_amdgcn_tensor_load_to_lds(g0, g1, z4, z4, 0);
#endif
    __builtin_amdgcn_s_wait_tensorcnt(0);
  }
  __syncthreads();
  xf = (const float*)(smem + OFF_Q);   // read staged tile from LDS
#else
  xf = xb;                             // fallback: read straight from global
#endif

  // ---- Phase 1: per-row L2 norm, write xh & nxh f16 to LDS ----
  {
    const int r  = tid >> 2;
    const int cb = (tid & 3) << 6;            // 4 lanes per row, 64 cols each
    float4 vbuf[16];
    float s = 0.f;
#pragma unroll
    for (int j = 0; j < 16; ++j) {
      vbuf[j] = *(const float4*)(xf + r * 256 + cb + j * 4);
      s += vbuf[j].x * vbuf[j].x + vbuf[j].y * vbuf[j].y +
           vbuf[j].z * vbuf[j].z + vbuf[j].w * vbuf[j].w;
    }
    s += __shfl_xor(s, 1, 32);
    s += __shfl_xor(s, 2, 32);
    const float inv = 1.0f / fmaxf(sqrtf(s), 1e-12f);
#pragma unroll
    for (int j = 0; j < 16; ++j) {
      const float4 v = vbuf[j];
      const int base = r * 256 + cb + j * 4;
      xh[base + 0] = (_Float16)v.x;  nxh[base + 0] = (_Float16)(v.x * inv);
      xh[base + 1] = (_Float16)v.y;  nxh[base + 1] = (_Float16)(v.y * inv);
      xh[base + 2] = (_Float16)v.z;  nxh[base + 2] = (_Float16)(v.z * inv);
      xh[base + 3] = (_Float16)v.w;  nxh[base + 3] = (_Float16)(v.w * inv);
    }
  }
  __syncthreads();   // also fences the transient x-f32 region before Q/K overwrite

  // ---- Phase 2: Q/K/V projections (per-head, WMMA f16) ----
  project_slice<PREW>(nxh, qw, wf + 0 * WMAT_HALFS, qb,  wsl, qh + w * 2048, w, lane, false);
  project_slice<PREW>(nxh, kw, wf + 1 * WMAT_HALFS, kbv, wsl, kh + w * 2048, w, lane, false);
  project_slice<PREW>(xh,  vw, wf + 2 * WMAT_HALFS, vb,  wsl, vt + w * 2048, w, lane, true);

  // ---- Phase 3a: S = QK^T * scale + mask, softmax, P -> per-wave LDS ----
  _Float16* P = wsl;                           // reuse staging: [64][64] f16
  const _Float16* qhh = qh + w * 2048;
  const _Float16* khh = kh + w * 2048;
  for (int mt = 0; mt < 4; ++mt) {
    v8f S[4];
    const v16h a = ldA(qhh + (mt * 16 + lr) * 32, 0, g);   // K=32: one chunk
#pragma unroll
    for (int nt = 0; nt < 4; ++nt) {
      v8f z = {};
      S[nt] = __builtin_amdgcn_wmma_f32_16x16x32_f16(false, a, false,
                  ldB(khh + (nt * 16 + lr) * 32, 0, g), (short)0, z, false, false);
    }
#pragma unroll
    for (int nt = 0; nt < 4; ++nt)
#pragma unroll
      for (int i = 0; i < 8; ++i) {
        const int row = mt * 16 + i + g * 8;
        const int col = nt * 16 + lr;
        S[nt][i] = S[nt][i] * SCALEF + mk[row * 64 + col];
      }
#pragma unroll
    for (int i = 0; i < 8; ++i) {
      float mx = fmaxf(fmaxf(S[0][i], S[1][i]), fmaxf(S[2][i], S[3][i]));
#pragma unroll
      for (int d = 1; d < 16; d <<= 1) mx = fmaxf(mx, __shfl_xor(mx, d, 32));
      float sm = 0.f;
#pragma unroll
      for (int nt = 0; nt < 4; ++nt) {
        const float e = __expf(S[nt][i] - mx);
        S[nt][i] = e; sm += e;
      }
#pragma unroll
      for (int d = 1; d < 16; d <<= 1) sm += __shfl_xor(sm, d, 32);
      const float rinv = 1.0f / sm;
#pragma unroll
      for (int nt = 0; nt < 4; ++nt)
        P[(mt * 16 + i + g * 8) * 64 + nt * 16 + lr] = (_Float16)(S[nt][i] * rinv);
    }
  }
  __syncthreads();   // all heads done with Q before overwriting it with O

  // ---- Phase 3b: O = P @ V  ->  oh[64][256] (reuses Q buffer) ----
  _Float16* oh = qh;
  const _Float16* vth = vt + w * 2048;
  for (int mt = 0; mt < 4; ++mt) {
    const _Float16* arow = P + (mt * 16 + lr) * 64;
#pragma unroll
    for (int nt = 0; nt < 2; ++nt) {
      v8f acc = {};
      const _Float16* brow = vth + (nt * 16 + lr) * 64;   // V^T row = head-dim
#pragma unroll
      for (int kk = 0; kk < 2; ++kk)
        acc = __builtin_amdgcn_wmma_f32_16x16x32_f16(false, ldA(arow, kk * 32, g),
                  false, ldB(brow, kk * 32, g), (short)0, acc, false, false);
#pragma unroll
      for (int i = 0; i < 8; ++i) {
        const int row = mt * 16 + i + g * 8;
        oh[row * 256 + w * 32 + nt * 16 + lr] = (_Float16)acc[i];
      }
    }
  }
  __syncthreads();   // O complete before cross-head final projection

  // ---- Phase 4: Y = O @ (pw*gain)^T + pb, wave covers cols [w*32, w*32+32) ----
  stage_weights<PREW>(pw, wf + 3 * WMAT_HALFS, wsl, w, lane);
  float* ob = out + (size_t)b * NTOK * CDIM;
  for (int mt = 0; mt < 4; ++mt) {
    v16h A[8];
    const _Float16* arow = oh + (mt * 16 + lr) * 256;
#pragma unroll
    for (int kk = 0; kk < 8; ++kk) A[kk] = ldA(arow, kk * 32, g);
#pragma unroll
    for (int nt = 0; nt < 2; ++nt) {
      v8f acc = {};
      const _Float16* brow = wsl + (nt * 16 + lr) * 256;
#pragma unroll
      for (int kk = 0; kk < 8; ++kk)
        acc = __builtin_amdgcn_wmma_f32_16x16x32_f16(false, A[kk], false,
                  ldB(brow, kk * 32, g), (short)0, acc, false, false);
      const float bv = pb[w * 32 + nt * 16 + lr];
#pragma unroll
      for (int i = 0; i < 8; ++i) {
        const int row = mt * 16 + i + g * 8;
        ob[row * 256 + w * 32 + nt * 16 + lr] = acc[i] + bv;
      }
    }
  }
}

extern "C" void kernel_launch(void* const* d_in, const int* in_sizes, int n_in,
                              void* d_out, int out_size, void* d_ws, size_t ws_size,
                              hipStream_t stream) {
  const float* x   = (const float*)d_in[0];
  const float* msk = (const float*)d_in[1];
  const float* qw  = (const float*)d_in[2];
  const float* qb  = (const float*)d_in[3];
  const float* kw  = (const float*)d_in[4];
  const float* kb  = (const float*)d_in[5];
  const float* vw  = (const float*)d_in[6];
  const float* vb  = (const float*)d_in[7];
  const float* pw  = (const float*)d_in[8];
  const float* pb  = (const float*)d_in[9];
  float* out = (float*)d_out;

  if (d_ws != nullptr && ws_size >= WS_NEEDED) {
    _Float16* wf = (_Float16*)d_ws;
    prep_weights_kernel<<<dim3(64), dim3(256), 0, stream>>>(qw, kw, vw, pw, wf);
    win_attn_kernel<true><<<dim3(4096), dim3(256), SMEM_BYTES, stream>>>(
        x, msk, qw, qb, kw, kb, vw, vb, pw, pb, wf, out);
  } else {
    win_attn_kernel<false><<<dim3(4096), dim3(256), SMEM_BYTES, stream>>>(
        x, msk, qw, qb, kw, kb, vw, vb, pw, pb, nullptr, out);
  }
}
